// LiGRU_48988396978776
// MI455X (gfx1250) — matrix operand order repaired and checked by reading, not compiled, not run on hardware
//
#include <hip/hip_runtime.h>
#include <math.h>

typedef __attribute__((ext_vector_type(16))) __bf16   v16bf;
typedef __attribute__((ext_vector_type(8)))  float    v8f;
typedef __attribute__((ext_vector_type(8)))  unsigned v8u;

#define B_  64
#define T_  4096
#define I_  64
#define H_  256
#define O_  64

__device__ __forceinline__ unsigned short f2bf_bits(float f) {
    return __builtin_bit_cast(unsigned short, (__bf16)f);
}
__device__ __forceinline__ unsigned pack2bf(float lo, float hi) {
    return (unsigned)f2bf_bits(lo) | ((unsigned)f2bf_bits(hi) << 16);
}
__device__ __forceinline__ float bf2f(unsigned short u) {
    return (float)__builtin_bit_cast(__bf16, u);
}
__device__ __forceinline__ v8f wmma_bf16(v16bf a, v16bf b, v8f c) {
    // (neg_a, A, neg_b, B, c_mod, C, reuse_a, reuse_b)
    return __builtin_amdgcn_wmma_f32_16x16x32_bf16(false, a, false, b, (short)0, c,
                                                   false, false);
}

__device__ __forceinline__ float fast_tanh(float x) {
#if __has_builtin(__builtin_amdgcn_tanhf)
    return __builtin_amdgcn_tanhf(x);
#elif __has_builtin(__builtin_amdgcn_tanh_f32)
    return __builtin_amdgcn_tanh_f32(x);
#else
    return tanhf(x);
#endif
}
__device__ __forceinline__ float fast_sigmoid(float x) {
#if __has_builtin(__builtin_amdgcn_exp2f) && __has_builtin(__builtin_amdgcn_rcpf)
    return __builtin_amdgcn_rcpf(1.0f + __builtin_amdgcn_exp2f(-1.44269504f * x));
#else
    return 1.0f / (1.0f + __expf(-x));
#endif
}

// gfx1250 async copy global->LDS (tracked by ASYNCcnt), per-lane 16B.
__device__ __forceinline__ void async_load_b128(unsigned lds_byte_addr,
                                                unsigned long long gaddr) {
    asm volatile("global_load_async_to_lds_b128 %0, %1, off"
                 :: "v"(lds_byte_addr), "v"(gaddr) : "memory");
}
__device__ __forceinline__ void wait_async_le2() {
    asm volatile("s_wait_asynccnt 0x2" ::: "memory");
}
__device__ __forceinline__ unsigned lds_off(const void* p) {
    // flat LDS address low 32 bits == wave LDS byte offset
    return (unsigned)(uintptr_t)p;
}

// ---------------------------------------------------------------------------
// ws layout (bf16), t-major so the recurrent kernel's per-step slice is one
// contiguous 512B block per (wave, matrix):
//   elem(t, b, hcol) = t*16384 + (b>>4)*4096 + (hcol>>4)*256 + (b&15)*16 + (hcol&15)
// ---------------------------------------------------------------------------

// Kernel 1: xz = x@Wz+bz, xh = x@Wh+bh for all (b,t).  bf16 WMMA, f32 acc.
__global__ __launch_bounds__(256) void ligru_proj(
    const float* __restrict__ x,  const float* __restrict__ Wz,
    const float* __restrict__ bz, const float* __restrict__ Wh,
    const float* __restrict__ bh, unsigned short* __restrict__ xzw,
    unsigned short* __restrict__ xhw)
{
    // 64 B-operands (2 mats * 16 n-tiles * 2 k-blocks), each 32 lanes * 8 dwords
    __shared__ unsigned sB[64 * 256];
    const int tid = threadIdx.x;
    for (int e = tid; e < 64 * 256; e += 256) {
        int v    = e & 7;
        int ln   = (e >> 3) & 31;
        int opi  = e >> 8;          // 0..63
        int mat  = opi >> 5;        // 0:Wz 1:Wh
        int rem  = opi & 31;
        int n    = rem >> 1;
        int kb   = rem & 1;
        int hf   = ln >> 4;
        int col  = (n << 4) + (ln & 15);
        int k0   = (kb << 5) + (hf << 4) + (v << 1);
        const float* W = mat ? Wh : Wz;
        sB[opi * 256 + ln * 8 + v] = pack2bf(W[k0 * H_ + col], W[(k0 + 1) * H_ + col]);
    }
    __syncthreads();

    const int lane = tid & 31, wave = tid >> 5;
    const int half = lane >> 4, colL = lane & 15;
    const int mt  = blockIdx.x * 8 + wave;   // 16-row M tile over B*T flat rows
    const int b   = mt >> 8;                 // T/16 = 256 t-tiles per batch row
    const int tb  = mt & 255;
    const int row = (mt << 4) + colL;        // lane's A row (M = lane%16)

    v16bf a[2];
    for (int kb = 0; kb < 2; ++kb) {
        v8u au;
        for (int v = 0; v < 8; ++v) {
            int kk = (kb << 5) + (half << 4) + (v << 1);
            const float2 f2 = *reinterpret_cast<const float2*>(x + row * I_ + kk);
            au[v] = pack2bf(f2.x, f2.y);
        }
        a[kb] = __builtin_bit_cast(v16bf, au);
    }

    for (int mat = 0; mat < 2; ++mat) {
        const float* bias   = mat ? bh  : bz;
        unsigned short* dst = mat ? xhw : xzw;
        for (int n = 0; n < 16; ++n) {
            float bv = bias[(n << 4) + colL];
            v8f acc = {bv, bv, bv, bv, bv, bv, bv, bv};
            for (int kb = 0; kb < 2; ++kb) {
                v8u bu = *reinterpret_cast<const v8u*>(
                    &sB[(((mat << 4) + n) * 2 + kb) * 256 + lane * 8]);
                acc = wmma_bf16(a[kb], __builtin_bit_cast(v16bf, bu), acc);
            }
            // t-major store: t = 16*tb + r + 8*half
            unsigned base = (unsigned)(tb << 4) * 16384u + (unsigned)(b >> 4) * 4096u
                          + (unsigned)n * 256u + (unsigned)(b & 15) * 16u + (unsigned)colL;
            for (int r = 0; r < 8; ++r)
                dst[base + (unsigned)(r + (half << 3)) * 16384u] = f2bf_bits(acc[r]);
        }
    }
}

// ---------------------------------------------------------------------------
// Kernel 2: recurrence + fused output projection.
// 4 WGs (16 batch rows each) * 640 threads = 20 waves:
//   waves 0..15 : own 16 H-cols; Uz/Uh slices persistent in VGPRs; per-step
//                 xz/xh slice prefetched one step ahead via async->LDS b128
//   waves 16..19: output projection (off the critical path)
// ---------------------------------------------------------------------------
__global__ __launch_bounds__(640) void ligru_rec(
    const float* __restrict__ h0,   const float* __restrict__ Uz,
    const float* __restrict__ Uh,   const float* __restrict__ Wout,
    const float* __restrict__ bout, const unsigned short* __restrict__ xzw,
    const unsigned short* __restrict__ xhw, float* __restrict__ out)
{
    __shared__ unsigned hbuf[2][2048];          // 2 x (16x256) bf16 h state, 16 KB
    __shared__ unsigned sW[32 * 256];           // W_out swizzled B-operands, 32 KB
    __shared__ unsigned short xls[2][2][4096];  // async x-proj staging, 32 KB

    const int tid  = threadIdx.x;
    const int lane = tid & 31, wave = tid >> 5;
    const int half = lane >> 4, colL = lane & 15;
    const int g = blockIdx.x;                   // batch rows 16g..16g+15

    for (int e = tid; e < 32 * 256; e += 640) {
        int v = e & 7, ln = (e >> 3) & 31, opi = e >> 8;
        int nb = opi >> 3, kb = opi & 7;
        int hf = ln >> 4, col = (nb << 4) + (ln & 15);
        int k0 = (kb << 5) + (hf << 4) + (v << 1);
        sW[opi * 256 + ln * 8 + v] = pack2bf(Wout[k0 * O_ + col], Wout[(k0 + 1) * O_ + col]);
    }

    v16bf uz[8], uh[8];
    v8f h_old;
    float bo = 0.f;

    auto issue_x = [&](int t, int pb) {
        size_t be = (size_t)t * 16384 + (size_t)g * 4096 + (size_t)wave * 256;
        unsigned long long gz =
            (unsigned long long)(uintptr_t)(xzw + be) + (unsigned)lane * 16u;
        unsigned long long gh =
            (unsigned long long)(uintptr_t)(xhw + be) + (unsigned)lane * 16u;
        unsigned lz = lds_off(&xls[pb][0][wave * 256]) + (unsigned)lane * 16u;
        unsigned lh = lds_off(&xls[pb][1][wave * 256]) + (unsigned)lane * 16u;
        async_load_b128(lz, gz);
        async_load_b128(lh, gh);
    };

    if (wave < 16) {
        const int w   = wave;
        const int col = (w << 4) + colL;        // owned H column
        for (int kb = 0; kb < 8; ++kb) {        // persistent U slices in VGPRs
            v8u az, ah;
            for (int v = 0; v < 8; ++v) {
                int k0 = (kb << 5) + (half << 4) + (v << 1);
                az[v] = pack2bf(Uz[k0 * H_ + col], Uz[(k0 + 1) * H_ + col]);
                ah[v] = pack2bf(Uh[k0 * H_ + col], Uh[(k0 + 1) * H_ + col]);
            }
            uz[kb] = __builtin_bit_cast(v16bf, az);
            uh[kb] = __builtin_bit_cast(v16bf, ah);
        }
        for (int r = 0; r < 8; ++r) {           // h0: f32 regs + bf16 LDS copy
            int m  = r + (half << 3);
            int bg = (g << 4) + m;
            float hv = h0[bg * H_ + col];
            h_old[r] = hv;
            ((unsigned short*)hbuf[0])[m * H_ + col] = f2bf_bits(hv);
        }
        issue_x(0, 0);                          // prime the async pipeline
    } else {
        bo = bout[((wave - 16) << 4) + colL];
    }
    __syncthreads();

    int p = 0;
    for (int t = 0; t < T_; ++t) {
        const int pb = t & 1;
        if (wave < 16) {
            // prefetch next step's slice, then wait for current slice (<=2 keeps
            // the just-issued pair in flight; in-order completion -> current done)
            const int tn = (t + 1 < T_) ? t + 1 : T_ - 1;
            issue_x(tn, pb ^ 1);
            wait_async_le2();

            v16bf a[8];
            for (int kb = 0; kb < 8; ++kb) {    // A = current h (bf16 LDS)
                v8u au;
                for (int v = 0; v < 8; ++v) {
                    int kk = (kb << 5) + (half << 4) + (v << 1);
                    au[v] = hbuf[p][colL * 128 + (kk >> 1)];
                }
                a[kb] = __builtin_bit_cast(v16bf, au);
            }
            const unsigned short* xz_s = &xls[pb][0][wave * 256];
            const unsigned short* xh_s = &xls[pb][1][wave * 256];
            v8f accZ, accH;
            for (int r = 0; r < 8; ++r) {
                int m = r + (half << 3);
                accZ[r] = bf2f(xz_s[m * 16 + colL]);
                accH[r] = bf2f(xh_s[m * 16 + colL]);
            }
            for (int kb = 0; kb < 8; ++kb) accZ = wmma_bf16(a[kb], uz[kb], accZ);
            for (int kb = 0; kb < 8; ++kb) accH = wmma_bf16(a[kb], uh[kb], accH);

            const int col = (wave << 4) + colL;
            for (int r = 0; r < 8; ++r) {
                float z  = fast_sigmoid(accZ[r]);
                float hc = fast_tanh(accH[r]);
                float hn = z * h_old[r] + (1.0f - z) * hc;
                h_old[r] = hn;
                ((unsigned short*)hbuf[p ^ 1])[(r + (half << 3)) * H_ + col] = f2bf_bits(hn);
            }
        }
        __syncthreads();
        if (wave >= 16) {
            const int nb = wave - 16;
            v8f acc = {bo, bo, bo, bo, bo, bo, bo, bo};
            for (int kb = 0; kb < 8; ++kb) {
                v8u au;
                for (int v = 0; v < 8; ++v) {
                    int kk = (kb << 5) + (half << 4) + (v << 1);
                    au[v] = hbuf[p ^ 1][colL * 128 + (kk >> 1)];
                }
                v8u bu = *reinterpret_cast<const v8u*>(&sW[(nb * 8 + kb) * 256 + lane * 8]);
                acc = wmma_bf16(__builtin_bit_cast(v16bf, au),
                                __builtin_bit_cast(v16bf, bu), acc);
            }
            for (int r = 0; r < 8; ++r) {
                unsigned o = ((unsigned)((g << 4) + r + (half << 3)) * T_ + (unsigned)t) * O_
                             + (nb << 4) + colL;
                out[o] = acc[r];
            }
        }
        p ^= 1;
    }
}

extern "C" void kernel_launch(void* const* d_in, const int* in_sizes, int n_in,
                              void* d_out, int out_size, void* d_ws, size_t ws_size,
                              hipStream_t stream) {
    const float* x    = (const float*)d_in[0];
    const float* h0   = (const float*)d_in[1];
    const float* Wz   = (const float*)d_in[2];
    const float* Uz   = (const float*)d_in[3];
    const float* bz   = (const float*)d_in[4];
    const float* Wh   = (const float*)d_in[5];
    const float* Uh   = (const float*)d_in[6];
    const float* bh   = (const float*)d_in[7];
    const float* Wout = (const float*)d_in[8];
    const float* bout = (const float*)d_in[9];
    float* out = (float*)d_out;

    unsigned short* xzw = (unsigned short*)d_ws;
    unsigned short* xhw = xzw + (size_t)T_ * 16384;   // 67,108,864 elems (128 MB)

    ligru_proj<<<dim3(2048), dim3(256), 0, stream>>>(x, Wz, bz, Wh, bh, xzw, xhw);
    ligru_rec<<<dim3(4), dim3(640), 0, stream>>>(h0, Uz, Uh, Wout, bout, xzw, xhw, out);
}
